// Attention_52493090292242
// MI455X (gfx1250) — matrix operand order repaired and checked by reading, not compiled, not run on hardware
//
#include <hip/hip_runtime.h>

// Problem constants (from reference): B=32, S=2048, H=1024.
#define BB 32
#define SS 2048
#define HH 1024
#define MBLOCK 128                 // rows of `value` staged per workgroup
#define LDS_ROW_BYTES 2064         // 1024 bf16 (2048B) + 16B pad to spread banks
#define LDS_A_BYTES (MBLOCK * LDS_ROW_BYTES)      // 264192 B
#define LDS_TOTAL   (LDS_A_BYTES + MBLOCK * 4)    // + 128-float score buffer

typedef __bf16 bf16_t;
typedef bf16_t v16bf __attribute__((ext_vector_type(16)));
typedef float  v8f   __attribute__((ext_vector_type(8)));

union FragBF {                     // 8 dwords <-> 16 bf16 (one WMMA operand)
    unsigned u[8];
    uint4    q[2];
    v16bf    v;
};

static __device__ __forceinline__ unsigned short f2bf(float f) {
    // round-to-nearest-even fp32 -> bf16
    unsigned u = __float_as_uint(f);
    unsigned r = u + 0x7FFFu + ((u >> 16) & 1u);
    return (unsigned short)(r >> 16);
}

static __device__ __forceinline__ float fast_tanh(float x) {
#if __has_builtin(__builtin_amdgcn_tanhf)
    return __builtin_amdgcn_tanhf(x);        // V_TANH_F32 (CDNA5 trans op)
#elif __has_builtin(__builtin_amdgcn_tanh_f32)
    return __builtin_amdgcn_tanh_f32(x);
#else
    return tanhf(x);
#endif
}

// ---------------------------------------------------------------------------
// Kernel 1: W2 [H,H] fp32 -> bf16 (same [o][h] row-major layout).
// ---------------------------------------------------------------------------
__global__ void cvt_w2_kernel(const float* __restrict__ W2,
                              unsigned short* __restrict__ W2bf) {
    int idx = blockIdx.x * 256 + threadIdx.x;       // one float4 per thread
    float4 d = ((const float4*)W2)[idx];
    uint2 p;
    p.x = (unsigned)f2bf(d.x) | ((unsigned)f2bf(d.y) << 16);
    p.y = (unsigned)f2bf(d.z) | ((unsigned)f2bf(d.w) << 16);
    ((uint2*)W2bf)[idx] = p;
}

// ---------------------------------------------------------------------------
// Kernel 2: out1[b][o] = sum_h key[b][h] * W1[o][h]   (tiny: 67 MFLOP, fp32)
// ---------------------------------------------------------------------------
__global__ void out1_kernel(const float* __restrict__ key,
                            const float* __restrict__ W1,
                            float* __restrict__ out1) {
    __shared__ float kb[HH];
    const int b    = blockIdx.x >> 2;
    const int oblk = blockIdx.x & 3;
    for (int i = threadIdx.x; i < HH; i += 256) kb[i] = key[b * HH + i];
    __syncthreads();
    const int o = oblk * 256 + threadIdx.x;
    const float4* w  = (const float4*)(W1 + (size_t)o * HH);
    const float4* k4 = (const float4*)kb;
    float s = 0.f;
    #pragma unroll 8
    for (int h = 0; h < HH / 4; ++h) {
        float4 wv = w[h], kv = k4[h];
        s += wv.x * kv.x + wv.y * kv.y + wv.z * kv.z + wv.w * kv.w;
    }
    out1[b * HH + o] = s;
}

// ---------------------------------------------------------------------------
// Kernel 3: fused scores[b,s] = sum_n v[n]*tanh(out1[b,n] + (value@W2^T)[b,s,n])
// Workgroup = 128 rows (one batch only: 128 | 2048). 8 waves.
// Each wave computes ALL 8 M-subtiles (128 rows) x ITS 8 N-tiles:
//   one B fragment feeds 8 WMMAs  -> 8x reuse, W2 L2 traffic = 1 GB total.
// Double-buffered A(LDS)/B(global) fragments software-pipeline the K loop.
// Cross-wave score reduction via ds_add_f32 into an LDS buffer.
// ---------------------------------------------------------------------------
__global__ __launch_bounds__(256, 1)
void attn_fused_kernel(const float* __restrict__ value,
                       const unsigned short* __restrict__ W2bf,
                       const float* __restrict__ out1,
                       const float* __restrict__ vvec,
                       float* __restrict__ scores) {
    extern __shared__ char smem[];
    float* sbuf = (float*)(smem + LDS_A_BYTES);     // 128 row scores
    const int tid  = threadIdx.x;
    const int base = blockIdx.x * MBLOCK;           // global row base

    // ---- Stage A: 128x1024 fp32 -> bf16 into LDS (coalesced b128 loads) ----
    const float4* src = (const float4*)(value + (size_t)base * HH);
    #pragma unroll 4
    for (int c = 0; c < (MBLOCK * HH / 4) / 256; ++c) {
        int f4   = c * 256 + tid;
        float4 d = src[f4];
        int row  = f4 >> 8;                         // 256 float4 per row
        int col4 = f4 & 255;
        uint2 p;
        p.x = (unsigned)f2bf(d.x) | ((unsigned)f2bf(d.y) << 16);
        p.y = (unsigned)f2bf(d.z) | ((unsigned)f2bf(d.w) << 16);
        *(uint2*)(smem + row * LDS_ROW_BYTES + col4 * 8) = p;
    }
    if (tid < MBLOCK) sbuf[tid] = 0.f;
    __syncthreads();

    const int wave     = tid >> 5;
    const int lane     = tid & 31;
    const int lane16   = lane & 15;
    const int laneHalf = lane >> 4;                 // 0: lanes 0-15, 1: 16-31
    const int b        = base / SS;                 // batch (uniform in block)

    // A fragment base addresses (per §7.12.2): lane holds row M=lane%16 of
    // subtile j, K runs [off,off+8) and [off+16,off+24), off = 8*laneHalf.
    const char* aBase[8];
    #pragma unroll
    for (int j = 0; j < 8; ++j)
        aBase[j] = smem + (j * 16 + lane16) * LDS_ROW_BYTES + laneHalf * 16;

#define LOAD_B(dst, kcIdx) do {                                           \
        const uint4* bp_ = (const uint4*)(bRow + (kcIdx) * 32);           \
        (dst).q[0] = bp_[0]; (dst).q[1] = bp_[2]; } while (0)

#define LOAD_A(dst, kcIdx) do {                                           \
        _Pragma("unroll")                                                 \
        for (int j_ = 0; j_ < 8; ++j_) {                                  \
            const char* ap_ = aBase[j_] + (kcIdx) * 64;                   \
            (dst)[j_].q[0] = *(const uint4*)ap_;                          \
            (dst)[j_].q[1] = *(const uint4*)(ap_ + 32); } } while (0)

#define WMMA_ALL(fa, fb) do {                                             \
        _Pragma("unroll")                                                 \
        for (int j_ = 0; j_ < 8; ++j_)                                    \
            acc[j_] = __builtin_amdgcn_wmma_f32_16x16x32_bf16(            \
                false, (fa)[j_].v, false, (fb).v, (short)0, acc[j_],      \
                false, false); } while (0)

    float sc[8][8];                 // running scores: row = j*16+laneHalf*8+i
    #pragma unroll
    for (int j = 0; j < 8; ++j)
        #pragma unroll
        for (int i = 0; i < 8; ++i) sc[j][i] = 0.f;

    // This wave's 8 N-tiles.
    for (int t = 0; t < 8; ++t) {
        const int nt   = wave * 8 + t;
        const int ncol = nt * 16 + lane16;          // output column of this lane
        // B fragment: lane holds column n=lane%16, K runs [off,off+8) and
        // [off+16,off+24), off = 8*laneHalf, in W2bf[n][k] layout.
        const unsigned short* bRow = W2bf + (size_t)ncol * HH + laneHalf * 8;
        if (t < 7)   // pull next N tile toward WGP$ (global_prefetch_b8)
            __builtin_prefetch(W2bf + (size_t)(ncol + 16) * HH, 0, 0);

        v8f acc[8];
        #pragma unroll
        for (int j = 0; j < 8; ++j)
            acc[j] = (v8f){0.f, 0.f, 0.f, 0.f, 0.f, 0.f, 0.f, 0.f};

        FragBF fa0[8], fa1[8], fb0, fb1;
        LOAD_B(fb0, 0);
        LOAD_A(fa0, 0);
        for (int kc = 0; kc < HH / 32; kc += 2) {   // software pipeline, x2
            LOAD_B(fb1, kc + 1);                    // stage kc+1 while kc runs
            LOAD_A(fa1, kc + 1);
            WMMA_ALL(fa0, fb0);
            if (kc + 2 < HH / 32) {                 // stage kc+2 while kc+1 runs
                LOAD_B(fb0, kc + 2);
                LOAD_A(fa0, kc + 2);
            }
            WMMA_ALL(fa1, fb1);
        }

        // Epilogue: C layout => lane has N=ncol; acc[j][i] is row j*16 +
        // i + 8*laneHalf. bias/vn are per-column scalars.
        const float bias = out1[b * HH + ncol];
        const float vn   = vvec[ncol];
        #pragma unroll
        for (int j = 0; j < 8; ++j)
            #pragma unroll
            for (int i = 0; i < 8; ++i)
                sc[j][i] += fast_tanh(acc[j][i] + bias) * vn;
    }

    // Reduce over the 16 lanes sharing each row (xor masks stay in-half),
    // then cross-wave combine via LDS float atomics.
    #pragma unroll
    for (int m = 8; m >= 1; m >>= 1)
        #pragma unroll
        for (int j = 0; j < 8; ++j)
            #pragma unroll
            for (int i = 0; i < 8; ++i)
                sc[j][i] += __shfl_xor(sc[j][i], m, 32);
    if (lane16 == 0) {
        #pragma unroll
        for (int j = 0; j < 8; ++j)
            #pragma unroll
            for (int i = 0; i < 8; ++i)
                atomicAdd(&sbuf[j * 16 + laneHalf * 8 + i], sc[j][i]); // ds_add_f32
    }
    __syncthreads();
    if (tid < MBLOCK) scores[base + tid] = sbuf[tid];

#undef LOAD_B
#undef LOAD_A
#undef WMMA_ALL
}

// ---------------------------------------------------------------------------
// Host launcher. Inputs: key[B,H], value[B,S,H], W1[H,H], W2[H,H], v[H].
// Workspace: [0, 2MB)  W2 in bf16;  [2MB, 2MB+128KB)  out1 fp32.
// ---------------------------------------------------------------------------
extern "C" void kernel_launch(void* const* d_in, const int* in_sizes, int n_in,
                              void* d_out, int out_size, void* d_ws, size_t ws_size,
                              hipStream_t stream) {
    const float* key   = (const float*)d_in[0];
    const float* value = (const float*)d_in[1];
    const float* W1    = (const float*)d_in[2];
    const float* W2    = (const float*)d_in[3];
    const float* vvec  = (const float*)d_in[4];

    unsigned short* W2bf = (unsigned short*)d_ws;                       // 2 MB
    float* out1          = (float*)((char*)d_ws + (size_t)HH * HH * 2); // 128 KB
    float* scores        = (float*)d_out;                               // [B,S]

    static bool attr_set = false;
    if (!attr_set) {   // raise dynamic-LDS cap for the ~258KB tile (not a stream op)
        hipFuncSetAttribute((const void*)attn_fused_kernel,
                            hipFuncAttributeMaxDynamicSharedMemorySize, LDS_TOTAL);
        attr_set = true;
    }

    // 1) W2 -> bf16
    cvt_w2_kernel<<<(HH * HH / 4) / 256, 256, 0, stream>>>(W2, W2bf);
    // 2) out1 = key @ W1^T
    out1_kernel<<<BB * (HH / 256), 256, 0, stream>>>(key, W1, out1);
    // 3) fused GEMM + tanh + dot-with-v reduction
    attn_fused_kernel<<<(BB * SS) / MBLOCK, 256, LDS_TOTAL, stream>>>(
        value, W2bf, out1, vvec, scores);
}
